// ConsensusLoss_31112743092793
// MI455X (gfx1250) — matrix operand order repaired and checked by reading, not compiled, not run on hardware
//
#include <hip/hip_runtime.h>
#include <hip/hip_bf16.h>
#include <math.h>

// ---------------------------------------------------------------------------
// ConsensusLoss for MI455X (gfx1250): STFT-as-GEMM on v_wmma_f32_16x16x32_f16
// Round 3: 2x2 register tiling (32 frames x 32 bins per wave) -> 8 WMMA per
// K-chunk with 4 ds_load_b128 + 8 global_load_b128; B bytes/WMMA halved.
// ---------------------------------------------------------------------------
typedef _Float16 v16h __attribute__((ext_vector_type(16)));
typedef _Float16 v8h  __attribute__((ext_vector_type(8)));
typedef float    v8f  __attribute__((ext_vector_type(8)));

#define BQ 4
#define SQ 4
#define CQ 2
#define TQ 441000

// workspace accumulator slots: 0=sep 1=recon 2=stft2048 3=stft512 4=penalty
#define ACC_N 8

__device__ __forceinline__ float wave_reduce32(float v) {
  #pragma unroll
  for (int off = 16; off > 0; off >>= 1) v += __shfl_down(v, off, 32);
  return v;
}

__global__ void init_acc_kernel(float* acc) {
  if (threadIdx.x < ACC_N) acc[threadIdx.x] = 0.0f;
}

// ---------------------------------------------------------------------------
// DFT twiddle tables, f16, layout [n_padded][k]: a B-fragment slice
// (N = lane%16, K = group*16 + h) is one contiguous 32B load.
// ---------------------------------------------------------------------------
__global__ void dft_table_kernel(_Float16* __restrict__ cosT,
                                 _Float16* __restrict__ sinT,
                                 int N, int rows) {
  const size_t total = (size_t)rows * (size_t)N;
  const size_t stride = (size_t)gridDim.x * blockDim.x;
  for (size_t idx = (size_t)blockIdx.x * blockDim.x + threadIdx.x; idx < total; idx += stride) {
    const int n = (int)(idx / (size_t)N);
    const int k = (int)(idx - (size_t)n * N);
    const int ph = (int)(((long long)k * (long long)n) % (long long)N); // exact arg reduction
    const float ang = 6.283185307179586f * (float)ph / (float)N;
    cosT[idx] = (_Float16)cosf(ang);
    sinT[idx] = (_Float16)sinf(ang); // im sign irrelevant for |X|
  }
}

// ---------------------------------------------------------------------------
// sep = mean|pred-target|, recon = mean|sum_s pred - mix|
// ---------------------------------------------------------------------------
__global__ __launch_bounds__(256)
void sep_recon_kernel(const float* __restrict__ pred,
                      const float* __restrict__ target,
                      const float* __restrict__ mix,
                      float* __restrict__ acc) {
  __shared__ float s_sep[8], s_rec[8];
  float sep = 0.0f, rec = 0.0f;
  const size_t NTOT = (size_t)BQ * CQ * TQ;
  const size_t CT   = (size_t)CQ * TQ;
  const size_t SCT  = (size_t)SQ * CQ * TQ;
  const size_t stride = (size_t)gridDim.x * blockDim.x;
  for (size_t i = (size_t)blockIdx.x * blockDim.x + threadIdx.x; i < NTOT; i += stride) {
    const size_t b = i / CT;
    const size_t r = i - b * CT;
    const size_t base = b * SCT + r;
    float ps = 0.0f;
    #pragma unroll
    for (int s = 0; s < SQ; ++s) {
      const float p = pred[base + (size_t)s * CT];
      const float t = target[base + (size_t)s * CT];
      sep += fabsf(p - t);
      ps  += p;
    }
    rec += fabsf(ps - mix[i]);
  }
  sep = wave_reduce32(sep);
  rec = wave_reduce32(rec);
  const int wave = threadIdx.x >> 5, lane = threadIdx.x & 31;
  if (lane == 0) { s_sep[wave] = sep; s_rec[wave] = rec; }
  __syncthreads();
  if (threadIdx.x == 0) {
    float a = 0.0f, c = 0.0f;
    for (int w = 0; w < (int)(blockDim.x >> 5); ++w) { a += s_sep[w]; c += s_rec[w]; }
    atomicAdd(acc + 0, a);
    atomicAdd(acc + 1, c);
  }
}

// ---------------------------------------------------------------------------
// LDS frame tile: 32 frames x NFFT halves of windowed reflect-padded mono,
// rows padded to ROW = NFFT+8 halves -> lane stride == 4 words mod 64 banks
// -> per-lane ds_load_b128 A reads are bank-conflict free.
// ---------------------------------------------------------------------------
template<int NFFT>
__device__ __forceinline__ void fill_frames32(_Float16* __restrict__ fr,
                                              const float* __restrict__ sig, // signal base (C*T floats)
                                              const float* __restrict__ win,
                                              int m0) {
  constexpr int HOP = NFFT / 4;
  constexpr int PAD = NFFT / 2;
  constexpr int ROW = NFFT + 8;
  for (int idx = threadIdx.x; idx < 32 * NFFT; idx += blockDim.x) {
    const int m = idx / NFFT;          // power-of-two -> shift
    const int k = idx & (NFFT - 1);
    int j = (m0 + m) * HOP + k - PAD;
    if (j < 0) j = -j;                 // reflect left
    if (j >= TQ) j = 2 * TQ - 2 - j;   // reflect right
    __builtin_prefetch(sig + j + 1024, 0, 1);  // global_prefetch_b8
    fr[m * ROW + k] = (_Float16)(0.5f * (sig[j] + sig[(size_t)TQ + j]) * win[k]);
  }
}

// ---------------------------------------------------------------------------
// DFT over K for a 2x2 tile block (32 frames x 32 bins) per wave:
// acc[0..3] = {re,im}x{nt0,nt1} for frames [0,16), acc[4..7] for [16,32).
// Per chunk: 4 ds_load_b128 (A0,A1) + 8 global_load_b128 (B) + 8 WMMA.
// ---------------------------------------------------------------------------
template<int NFFT>
__device__ __forceinline__ void dft_2x2(const _Float16* __restrict__ fr,
                                        const _Float16* __restrict__ cosT,
                                        const _Float16* __restrict__ sinT,
                                        int ncol0, int ncol1, int mloc, int group,
                                        v8f (&acc)[8]) {
  constexpr int ROW = NFFT + 8;
  const _Float16* c0 = cosT + (size_t)ncol0 * NFFT + group * 16;
  const _Float16* s0 = sinT + (size_t)ncol0 * NFFT + group * 16;
  const _Float16* c1 = cosT + (size_t)ncol1 * NFFT + group * 16;
  const _Float16* s1 = sinT + (size_t)ncol1 * NFFT + group * 16;
  const _Float16* frow0 = fr + mloc * ROW + group * 16;           // frames 0..15
  const _Float16* frow1 = fr + (16 + mloc) * ROW + group * 16;    // frames 16..31
  for (int kb = 0; kb < NFFT; kb += 32) {
    // 16-bit A 16x32 layout (matching B): lane needs K = kb + group*16 + {0..15}
    // -> two aligned 16B LDS reads per fragment.
    const v8h a0lo = *(const v8h*)(frow0 + kb);
    const v8h a0hi = *(const v8h*)(frow0 + kb + 8);
    const v8h a1lo = *(const v8h*)(frow1 + kb);
    const v8h a1hi = *(const v8h*)(frow1 + kb + 8);
    const v16h a0 = __builtin_shufflevector(a0lo, a0hi, 0, 1, 2, 3, 4, 5, 6, 7,
                                            8, 9, 10, 11, 12, 13, 14, 15);
    const v16h a1 = __builtin_shufflevector(a1lo, a1hi, 0, 1, 2, 3, 4, 5, 6, 7,
                                            8, 9, 10, 11, 12, 13, 14, 15);
    const v16h bc0 = *(const v16h*)(c0 + kb);
    const v16h bs0 = *(const v16h*)(s0 + kb);
    const v16h bc1 = *(const v16h*)(c1 + kb);
    const v16h bs1 = *(const v16h*)(s1 + kb);
    acc[0] = __builtin_amdgcn_wmma_f32_16x16x32_f16(false, a0, false, bc0, (short)0, acc[0], false, false);
    acc[1] = __builtin_amdgcn_wmma_f32_16x16x32_f16(false, a0, false, bs0, (short)0, acc[1], false, false);
    acc[2] = __builtin_amdgcn_wmma_f32_16x16x32_f16(false, a0, false, bc1, (short)0, acc[2], false, false);
    acc[3] = __builtin_amdgcn_wmma_f32_16x16x32_f16(false, a0, false, bs1, (short)0, acc[3], false, false);
    acc[4] = __builtin_amdgcn_wmma_f32_16x16x32_f16(false, a1, false, bc0, (short)0, acc[4], false, false);
    acc[5] = __builtin_amdgcn_wmma_f32_16x16x32_f16(false, a1, false, bs0, (short)0, acc[5], false, false);
    acc[6] = __builtin_amdgcn_wmma_f32_16x16x32_f16(false, a1, false, bc1, (short)0, acc[6], false, false);
    acc[7] = __builtin_amdgcn_wmma_f32_16x16x32_f16(false, a1, false, bs1, (short)0, acc[7], false, false);
  }
}

// ---------------------------------------------------------------------------
// STFT L1 loss: sum |log1p|STFT(p)|| - log1p|STFT(t)|| into acc slot.
// grid = (frameBlocksOf32, pairGroupsOf4, 16 signals), block = 128 (4 waves);
// each wave owns 32 frames x bin tiles {nt0, nt0+1}.
// ---------------------------------------------------------------------------
template<int NFFT, int NFRAMES, int NBINS, int NT>
__global__ __launch_bounds__(128)
void stft_l1_kernel(const float* __restrict__ pred,
                    const float* __restrict__ target,
                    const float* __restrict__ win,
                    const _Float16* __restrict__ cosT,
                    const _Float16* __restrict__ sinT,
                    float* __restrict__ accslot) {
  extern __shared__ _Float16 fr[];

  const int tid  = threadIdx.x;
  const int lane = tid & 31;
  const int wave = tid >> 5;
  const int m0   = blockIdx.x * 32;
  const int nt0  = (blockIdx.y * 4 + wave) * 2;
  const int nt1c = (nt0 + 1 < NT) ? (nt0 + 1) : (NT - 1); // clamp, mask in epilogue
  const size_t sbase = (size_t)blockIdx.z * ((size_t)CQ * TQ);

  const int group = lane >> 4;
  const int mloc  = lane & 15;
  const bool live = (nt0 < NT);
  const int ncol0 = nt0 * 16 + mloc;
  const int ncol1 = nt1c * 16 + mloc;

  v8f acc[8];
  #pragma unroll
  for (int t = 0; t < 8; ++t) acc[t] = (v8f){};

  fill_frames32<NFFT>(fr, pred + sbase, win, m0);
  __syncthreads();
  if (live) dft_2x2<NFFT>(fr, cosT, sinT, ncol0, ncol1, mloc, group, acc);

  // fold pred pass into 4x log1p(mag) registers to free accumulators
  v8f lp[4];
  #pragma unroll
  for (int t = 0; t < 4; ++t) {
    #pragma unroll
    for (int r = 0; r < 8; ++r) {
      const float re = acc[2 * t][r], im = acc[2 * t + 1][r];
      lp[t][r] = log1pf(sqrtf(re * re + im * im));
    }
    acc[2 * t] = (v8f){};
    acc[2 * t + 1] = (v8f){};
  }

  __syncthreads();
  fill_frames32<NFFT>(fr, target + sbase, win, m0);
  __syncthreads();
  if (live) dft_2x2<NFFT>(fr, cosT, sinT, ncol0, ncol1, mloc, group, acc);

  float part = 0.0f;
  if (live) {
    #pragma unroll
    for (int t = 0; t < 4; ++t) {                  // t = mt*2 + nsel
      const int mt = t >> 1, nsel = t & 1;
      const int n = (nt0 + nsel) * 16 + mloc;
      const bool nok = (nsel == 0) ? (n < NBINS) : (nt0 + 1 < NT && n < NBINS);
      if (nok) {
        #pragma unroll
        for (int r = 0; r < 8; ++r) {
          const int m = m0 + 16 * mt + r + group * 8;
          if (m < NFRAMES) {
            const int ai = 4 * mt + 2 * nsel;
            const float re = acc[ai][r], im = acc[ai + 1][r];
            const float lt = log1pf(sqrtf(re * re + im * im));
            part += fabsf(lp[2 * mt + nsel][r] - lt);
          }
        }
      }
    }
  }
  part = wave_reduce32(part);
  if (lane == 0 && part != 0.0f) atomicAdd(accslot, part);
}

// ---------------------------------------------------------------------------
// Crosstalk penalty (n_fft=1024, pred only): sum over (b, frame, bin in [60,513))
// of 0.5*((sum_s mag)^2 - sum_s mag^2). Bin tiles 3..32 as 15 pairs.
// grid = (54 frameBlocksOf32, 4 pairGroups, B=4), block = 128
// ---------------------------------------------------------------------------
__global__ __launch_bounds__(128)
void penalty_kernel(const float* __restrict__ pred,
                    const float* __restrict__ win,
                    const _Float16* __restrict__ cosT,
                    const _Float16* __restrict__ sinT,
                    float* __restrict__ accslot) {
  constexpr int NFFT = 1024;
  constexpr int NFRAMES = 1723;
  extern __shared__ _Float16 fr[];

  const int tid  = threadIdx.x;
  const int lane = tid & 31;
  const int wave = tid >> 5;
  const int m0   = blockIdx.x * 32;
  const int q    = blockIdx.y * 4 + wave;   // pair index, 15 live
  const int nt0  = 3 + 2 * q;               // tiles 3..32 cover bins [48,528)
  const int b    = blockIdx.z;

  const int group = lane >> 4;
  const int mloc  = lane & 15;
  const bool live = (q < 15);
  const int nt0c  = live ? nt0 : 3;
  const int ncol0 = nt0c * 16 + mloc;
  const int ncol1 = (nt0c + 1) * 16 + mloc;

  v8f ssum[4], ssq[4];
  #pragma unroll
  for (int t = 0; t < 4; ++t) { ssum[t] = (v8f){}; ssq[t] = (v8f){}; }

  for (int s = 0; s < SQ; ++s) {
    const size_t sbase = ((size_t)(b * SQ + s)) * ((size_t)CQ * TQ);
    __syncthreads();                 // protect LDS from previous iteration's readers
    fill_frames32<NFFT>(fr, pred + sbase, win, m0);
    __syncthreads();
    if (live) {
      v8f acc[8];
      #pragma unroll
      for (int t = 0; t < 8; ++t) acc[t] = (v8f){};
      dft_2x2<NFFT>(fr, cosT, sinT, ncol0, ncol1, mloc, group, acc);
      #pragma unroll
      for (int t = 0; t < 4; ++t) {
        #pragma unroll
        for (int r = 0; r < 8; ++r) {
          const float re = acc[2 * t][r], im = acc[2 * t + 1][r];
          const float g = sqrtf(re * re + im * im);
          ssum[t][r] += g;
          ssq[t][r]  += g * g;
        }
      }
    }
  }

  float part = 0.0f;
  if (live) {
    #pragma unroll
    for (int t = 0; t < 4; ++t) {                  // t = mt*2 + nsel
      const int mt = t >> 1, nsel = t & 1;
      const int n = (nt0 + nsel) * 16 + mloc;
      if (n >= 60 && n < 513) {
        #pragma unroll
        for (int r = 0; r < 8; ++r) {
          const int m = m0 + 16 * mt + r + group * 8;
          if (m < NFRAMES) {
            const int ti = 2 * mt + nsel;
            part += 0.5f * (ssum[ti][r] * ssum[ti][r] - ssq[ti][r]);
          }
        }
      }
    }
  }
  part = wave_reduce32(part);
  if (lane == 0 && part != 0.0f) atomicAdd(accslot, part);
}

// ---------------------------------------------------------------------------
__global__ void combine_kernel(const float* __restrict__ acc, float* __restrict__ out) {
  if (threadIdx.x == 0 && blockIdx.x == 0) {
    const float sep  = acc[0] / 14112000.0f;                       // B*S*C*T
    const float rec  = acc[1] / 3528000.0f;                        // B*C*T
    const float stft = acc[2] / (16.0f * 862.0f * 1025.0f)         // n_fft=2048
                     + acc[3] / (16.0f * 3446.0f * 257.0f);        // n_fft=512
    const float pen  = acc[4] / (1723.0f * 453.0f * 24.0f);        // frames*bins / count
    out[0] = sep + rec + 0.1f * stft + 0.05f * pen;
  }
}

// ---------------------------------------------------------------------------
extern "C" void kernel_launch(void* const* d_in, const int* in_sizes, int n_in,
                              void* d_out, int out_size, void* d_ws, size_t ws_size,
                              hipStream_t stream) {
  const float* pred    = (const float*)d_in[0];
  const float* target  = (const float*)d_in[1];
  const float* mix     = (const float*)d_in[2];
  const float* win512  = (const float*)d_in[3];
  const float* win1024 = (const float*)d_in[4];
  const float* win2048 = (const float*)d_in[5];
  float* out = (float*)d_out;

  // workspace carve-up (accumulators + f16 twiddle tables, ~11.3 MB total)
  float* acc = (float*)d_ws;
  char* p = (char*)d_ws + 256;
  _Float16* cos2048 = (_Float16*)p; p += (size_t)1040 * 2048 * sizeof(_Float16);
  _Float16* sin2048 = (_Float16*)p; p += (size_t)1040 * 2048 * sizeof(_Float16);
  _Float16* cos1024 = (_Float16*)p; p += (size_t)528 * 1024 * sizeof(_Float16);
  _Float16* sin1024 = (_Float16*)p; p += (size_t)528 * 1024 * sizeof(_Float16);
  _Float16* cos512  = (_Float16*)p; p += (size_t)272 * 512 * sizeof(_Float16);
  _Float16* sin512  = (_Float16*)p; p += (size_t)272 * 512 * sizeof(_Float16);
  (void)ws_size; (void)in_sizes; (void)n_in; (void)out_size;

  init_acc_kernel<<<1, 32, 0, stream>>>(acc);

  dft_table_kernel<<<2048, 256, 0, stream>>>(cos2048, sin2048, 2048, 1040);
  dft_table_kernel<<<1024, 256, 0, stream>>>(cos1024, sin1024, 1024, 528);
  dft_table_kernel<<<512,  256, 0, stream>>>(cos512,  sin512,  512,  272);

  sep_recon_kernel<<<1536, 256, 0, stream>>>(pred, target, mix, acc);

  // n_fft=2048: 862 frames -> 27 blocks of 32; 65 bin tiles -> 33 pairs -> 9 groups
  stft_l1_kernel<2048, 862, 1025, 65>
      <<<dim3(27, 9, 16), 128, (size_t)32 * (2048 + 8) * sizeof(_Float16), stream>>>(
          pred, target, win2048, cos2048, sin2048, acc + 2);

  // n_fft=512: 3446 frames -> 108 blocks of 32; 17 bin tiles -> 9 pairs -> 3 groups
  stft_l1_kernel<512, 3446, 257, 17>
      <<<dim3(108, 3, 16), 128, (size_t)32 * (512 + 8) * sizeof(_Float16), stream>>>(
          pred, target, win512, cos512, sin512, acc + 3);

  // penalty: n_fft=1024, 1723 frames -> 54 blocks of 32; bin tiles 3..32 -> 15 pairs
  penalty_kernel<<<dim3(54, 4, 4), 128, (size_t)32 * (1024 + 8) * sizeof(_Float16), stream>>>(
      pred, win1024, cos1024, sin1024, acc + 4);

  combine_kernel<<<1, 32, 0, stream>>>(acc, out);
}
